// MultiHeadAttention_83305185673435
// MI455X (gfx1250) — compile-verified, matrix operand order checked by reading, not run on hardware
//
#include <hip/hip_runtime.h>
#include <hip/hip_bf16.h>

// MI455X / gfx1250: wave32, WMMA 16x16x32 bf16 -> f32 accumulate.
typedef __attribute__((ext_vector_type(16))) __bf16 v16bf;
typedef __attribute__((ext_vector_type(8)))  float  v8f;
typedef __attribute__((ext_vector_type(4)))  unsigned int u32x4;

#define D_MODEL 1024
#define SEQ     2048
#define NH      16
#define DK      64

union Frag16 {
  v16bf v;
  u32x4 u[2];
};

// Load one 16x32 bf16 WMMA fragment from a row-major LDS tile.
// A-fragment: lane = row (lanes 16-31 hold the other K-half of the same rows).
// B-fragment: identical pattern on the transposed tile (lane = output column).
// Per-lane: elements 0..7  = k in [hi*8, hi*8+8)      -> 16B at +hi*16
//           elements 8..15 = k in [16+hi*8, 16+hi*8+8) -> 16B at +32+hi*16
__device__ inline v16bf lds_load_frag(const __bf16* tile, int row0, int koff, int stride_e) {
  const int lane = threadIdx.x & 31;
  const int r  = lane & 15;
  const int hi = lane >> 4;
  const char* p = (const char*)(tile + (size_t)(row0 + r) * stride_e + koff);
  Frag16 f;
  f.u[0] = *(const u32x4*)(p + hi * 16);
  f.u[1] = *(const u32x4*)(p + 32 + hi * 16);
  return f.v;
}

__device__ inline v8f wmma_bf16(v16bf a, v16bf b, v8f c) {
  return __builtin_amdgcn_wmma_f32_16x16x32_bf16(false, a, false, b, (short)0, c, false, false);
}

// ---------------------------------------------------------------------------
// Weight prep: W[k][n] f32 -> Wt[n][k] bf16 (so GEMM B-tiles are row loads).
// ---------------------------------------------------------------------------
__global__ __launch_bounds__(256) void transpose_cvt_bf16(const float* __restrict__ W,
                                                          __bf16* __restrict__ Wt) {
  __shared__ float tile[32][33];
  const int bx = blockIdx.x * 32;  // n
  const int by = blockIdx.y * 32;  // k
  const int tx = threadIdx.x, ty = threadIdx.y;  // block (32,8)
  for (int i = 0; i < 4; ++i)
    tile[ty + i * 8][tx] = W[(size_t)(by + ty + i * 8) * D_MODEL + bx + tx];
  __syncthreads();
  for (int i = 0; i < 4; ++i)
    Wt[(size_t)(bx + ty + i * 8) * D_MODEL + by + tx] = (__bf16)tile[tx][ty + i * 8];
}

// ---------------------------------------------------------------------------
// GEMM: out = scale * (A[8192x1024] @ Wt^T + bias)
//   A_IS_F32: A is f32 (converted to bf16 into LDS) else bf16.
//   OUT_MODE 0: bf16, head-major [B,H,S,64]    (Q, K)
//   OUT_MODE 1: bf16, head-major transposed [B,H,64,S]  (V^T)
//   OUT_MODE 2: f32, row-major [8192,1024]     (final output)
// Block 128x128x(BK=32), 256 threads / 8 waves, wave tile 32x64 -> 8 WMMA/K-step.
// ---------------------------------------------------------------------------
template <int A_IS_F32, int OUT_MODE>
__global__ __launch_bounds__(256) void gemm_wmma(const void* __restrict__ Aptr,
                                                 const __bf16* __restrict__ Bt,
                                                 const float* __restrict__ bias,
                                                 void* __restrict__ out, float scale) {
  constexpr int LDT = 40;  // LDS stride (bf16): 80B -> 16B-aligned b128, minimal conflicts
  __shared__ __align__(16) __bf16 As[128 * LDT];
  __shared__ __align__(16) __bf16 Bs[128 * LDT];

  const int tid  = threadIdx.x;
  const int lane = tid & 31;
  const int w  = tid >> 5;
  const int wm = w >> 1, wn = w & 1;
  const int m0 = blockIdx.y * 128;
  const int n0 = blockIdx.x * 128;

  v8f acc[2][4];
  for (int i = 0; i < 2; ++i)
    for (int j = 0; j < 4; ++j)
      for (int e = 0; e < 8; ++e) acc[i][j][e] = 0.f;

  for (int kt = 0; kt < D_MODEL / 32; ++kt) {
    const int k0 = kt * 32;
    __syncthreads();
    if (A_IS_F32) {
      const float* A = (const float*)Aptr;
      for (int p = 0; p < 4; ++p) {
        int idx = p * 256 + tid;
        int row = idx >> 3, cg = idx & 7;
        float4 x4 = *(const float4*)(A + (size_t)(m0 + row) * D_MODEL + k0 + cg * 4);
        __bf16* d = As + row * LDT + cg * 4;
        d[0] = (__bf16)x4.x; d[1] = (__bf16)x4.y; d[2] = (__bf16)x4.z; d[3] = (__bf16)x4.w;
      }
    } else {
      const __bf16* A = (const __bf16*)Aptr;
      for (int p = 0; p < 2; ++p) {
        int idx = p * 256 + tid;
        int row = idx >> 2, cg = idx & 3;
        *(u32x4*)(As + row * LDT + cg * 8) =
            *(const u32x4*)(A + (size_t)(m0 + row) * D_MODEL + k0 + cg * 8);
      }
    }
    for (int p = 0; p < 2; ++p) {
      int idx = p * 256 + tid;
      int row = idx >> 2, cg = idx & 3;
      *(u32x4*)(Bs + row * LDT + cg * 8) =
          *(const u32x4*)(Bt + (size_t)(n0 + row) * D_MODEL + k0 + cg * 8);
    }
    if (kt + 1 < D_MODEL / 32)  // hint next B k-slab into cache
      __builtin_prefetch(Bt + (size_t)(n0 + (tid >> 1)) * D_MODEL + k0 + 32, 0, 1);
    __syncthreads();

    v16bf a0 = lds_load_frag(As, wm * 32, 0, LDT);
    v16bf a1 = lds_load_frag(As, wm * 32 + 16, 0, LDT);
    for (int j = 0; j < 4; ++j) {
      v16bf b = lds_load_frag(Bs, wn * 64 + j * 16, 0, LDT);
      acc[0][j] = wmma_bf16(a0, b, acc[0][j]);
      acc[1][j] = wmma_bf16(a1, b, acc[1][j]);
    }
  }

  // Epilogue: bias + scale + layout scatter.
  const int r15 = lane & 15, hi = lane >> 4;
  for (int j = 0; j < 4; ++j) {
    const int n = n0 + wn * 64 + j * 16 + r15;
    const float bv = bias[n];
    const int h = n >> 6, d = n & 63;
    for (int i = 0; i < 2; ++i) {
      for (int r = 0; r < 8; ++r) {
        const int m = m0 + wm * 32 + i * 16 + r + hi * 8;
        const float val = (acc[i][j][r] + bv) * scale;
        if (OUT_MODE == 2) {
          ((float*)out)[(size_t)m * D_MODEL + n] = val;
        } else {
          const int b = m >> 11, s = m & (SEQ - 1);
          size_t addr;
          if (OUT_MODE == 0)
            addr = ((size_t)(b * NH + h) * SEQ + s) * DK + d;
          else
            addr = ((size_t)(b * NH + h) * DK + d) * SEQ + s;
          ((__bf16*)out)[addr] = (__bf16)val;
        }
      }
    }
  }
}

// ---------------------------------------------------------------------------
// Flash attention: grid (S/64, B*H), 128 threads = 4 waves x 16 q-rows.
// Q is pre-scaled by 1/sqrt(dk). Online softmax fully in registers:
// WMMA C-layout (vgpr r = row, lanes stripe columns) lets a 16-lane
// __shfl_xor tree deliver row max/sum to exactly the lanes holding that row.
// ---------------------------------------------------------------------------
__global__ __launch_bounds__(128) void attention_wmma(const __bf16* __restrict__ Q,
                                                      const __bf16* __restrict__ K,
                                                      const __bf16* __restrict__ Vt,
                                                      __bf16* __restrict__ O) {
  constexpr int LQ = 72;   // stride for 64-wide-K tiles (144B, 16B aligned)
  constexpr int LP = 136;  // stride for 128-wide-K tiles (272B, 16B aligned)
  __shared__ __align__(16) __bf16 Qs[64 * LQ];    // 64 q-rows x 64 d
  __shared__ __align__(16) __bf16 Ks[128 * LQ];   // 128 keys x 64 d   (= B^T of K^T)
  __shared__ __align__(16) __bf16 Vts[64 * LP];   // 64 d x 128 keys   (= B^T of V)
  __shared__ __align__(16) __bf16 Ps[64 * LP];    // probs, A-fragment staging

  const int tid  = threadIdx.x;
  const int lane = tid & 31;
  const int w    = tid >> 5;
  const int r0   = w * 16;
  const int qt   = blockIdx.x;
  const int bh   = blockIdx.y;
  const int r15  = lane & 15, hi = lane >> 4;

  const size_t qbase = ((size_t)bh * SEQ + qt * 64) * DK;
  for (int p = 0; p < 4; ++p) {
    int idx = p * 128 + tid;
    int row = idx >> 3, cg = idx & 7;
    *(u32x4*)(Qs + row * LQ + cg * 8) =
        *(const u32x4*)(Q + qbase + (size_t)row * DK + cg * 8);
  }
  __syncthreads();
  const v16bf qf0 = lds_load_frag(Qs, r0, 0, LQ);   // loop-invariant Q fragments
  const v16bf qf1 = lds_load_frag(Qs, r0, 32, LQ);

  v8f accO[4];
  for (int j = 0; j < 4; ++j)
    for (int e = 0; e < 8; ++e) accO[j][e] = 0.f;
  float mrow[8], lrow[8];
  for (int r = 0; r < 8; ++r) { mrow[r] = -1e30f; lrow[r] = 0.f; }

  const int nkb = ((qt * 64 + 63) >> 7) + 1;  // causal: key blocks needed

  for (int kb = 0; kb < nkb; ++kb) {
    __syncthreads();  // previous P@V done before overwriting Ks/Vts
    for (int p = 0; p < 8; ++p) {  // Ks: 128 x 64
      int idx = p * 128 + tid;
      int row = idx >> 3, cg = idx & 7;
      *(u32x4*)(Ks + row * LQ + cg * 8) =
          *(const u32x4*)(K + ((size_t)bh * SEQ + kb * 128 + row) * DK + cg * 8);
    }
    for (int p = 0; p < 8; ++p) {  // Vts: 64 x 128
      int idx = p * 128 + tid;
      int row = idx >> 4, cg = idx & 15;
      *(u32x4*)(Vts + row * LP + cg * 8) =
          *(const u32x4*)(Vt + ((size_t)bh * DK + row) * SEQ + kb * 128 + cg * 8);
    }
    __syncthreads();

    // Scores: 16 q-rows x 128 keys per wave (16 WMMAs)
    float sc[8][8];
    for (int n = 0; n < 8; ++n) {
      v16bf kf0 = lds_load_frag(Ks, n * 16, 0, LQ);
      v16bf kf1 = lds_load_frag(Ks, n * 16, 32, LQ);
      v8f s;
      for (int e = 0; e < 8; ++e) s[e] = 0.f;
      s = wmma_bf16(qf0, kf0, s);
      s = wmma_bf16(qf1, kf1, s);
      for (int e = 0; e < 8; ++e) sc[n][e] = s[e];
    }
    if (kb == nkb - 1) {  // causal mask only on the diagonal block
      for (int n = 0; n < 8; ++n) {
        const int col = kb * 128 + n * 16 + r15;
        for (int r = 0; r < 8; ++r) {
          const int rowq = qt * 64 + r0 + r + hi * 8;
          if (col > rowq) sc[n][r] = -1e30f;
        }
      }
    }
    // Online softmax update (per row; reductions over 16-lane halves)
    for (int r = 0; r < 8; ++r) {
      float mx = sc[0][r];
      for (int n = 1; n < 8; ++n) mx = fmaxf(mx, sc[n][r]);
      for (int off = 1; off < 16; off <<= 1) mx = fmaxf(mx, __shfl_xor(mx, off));
      const float mnew  = fmaxf(mrow[r], mx);
      const float alpha = __expf(mrow[r] - mnew);
      mrow[r] = mnew;
      float rs = 0.f;
      for (int n = 0; n < 8; ++n) {
        const float pe = __expf(sc[n][r] - mnew);
        sc[n][r] = pe;
        rs += pe;
      }
      for (int off = 1; off < 16; off <<= 1) rs += __shfl_xor(rs, off);
      lrow[r] = lrow[r] * alpha + rs;
      for (int j = 0; j < 4; ++j) accO[j][r] *= alpha;
    }
    // Stage P as bf16 for A-fragment reload
    for (int n = 0; n < 8; ++n)
      for (int r = 0; r < 8; ++r)
        Ps[(size_t)(r0 + r + hi * 8) * LP + n * 16 + r15] = (__bf16)sc[n][r];
    __syncthreads();

    // O += P @ V (16 WMMAs)
    for (int kk = 0; kk < 4; ++kk) {
      v16bf pf = lds_load_frag(Ps, r0, kk * 32, LP);
      for (int j = 0; j < 4; ++j) {
        v16bf vf = lds_load_frag(Vts, j * 16, kk * 32, LP);
        accO[j] = wmma_bf16(pf, vf, accO[j]);
      }
    }
  }

  // O /= l, write bf16 head-merged [B,S,1024] for the output projection
  const int b = bh >> 4, h = bh & 15;
  for (int r = 0; r < 8; ++r) {
    const float inv = 1.0f / lrow[r];
    const int s = qt * 64 + r0 + r + hi * 8;
    const size_t base = ((size_t)b * SEQ + s) * D_MODEL + h * DK;
    for (int j = 0; j < 4; ++j)
      O[base + j * 16 + r15] = (__bf16)(accO[j][r] * inv);
  }
}

// ---------------------------------------------------------------------------
extern "C" void kernel_launch(void* const* d_in, const int* in_sizes, int n_in,
                              void* d_out, int out_size, void* d_ws, size_t ws_size,
                              hipStream_t stream) {
  (void)in_sizes; (void)n_in; (void)out_size; (void)ws_size;
  const float* x  = (const float*)d_in[0];
  const float* Wq = (const float*)d_in[1];
  const float* bq = (const float*)d_in[2];
  const float* Wk = (const float*)d_in[3];
  const float* bk = (const float*)d_in[4];
  const float* Wv = (const float*)d_in[5];
  const float* bv = (const float*)d_in[6];
  const float* Wo = (const float*)d_in[7];
  const float* bo = (const float*)d_in[8];

  char* ws = (char*)d_ws;
  const size_t WB  = (size_t)D_MODEL * D_MODEL * sizeof(__bf16);  // 2 MB
  const size_t QKV = (size_t)4 * SEQ * D_MODEL * sizeof(__bf16);  // 16 MB
  __bf16* Wqt = (__bf16*)(ws + 0 * WB);
  __bf16* Wkt = (__bf16*)(ws + 1 * WB);
  __bf16* Wvt = (__bf16*)(ws + 2 * WB);
  __bf16* Wot = (__bf16*)(ws + 3 * WB);
  __bf16* Qb  = (__bf16*)(ws + 4 * WB);
  __bf16* Kb  = (__bf16*)(ws + 4 * WB + QKV);
  __bf16* Vtb = (__bf16*)(ws + 4 * WB + 2 * QKV);
  __bf16* Ob  = (__bf16*)(ws + 4 * WB + 3 * QKV);

  dim3 tb(32, 8), tg(32, 32);
  transpose_cvt_bf16<<<tg, tb, 0, stream>>>(Wq, Wqt);
  transpose_cvt_bf16<<<tg, tb, 0, stream>>>(Wk, Wkt);
  transpose_cvt_bf16<<<tg, tb, 0, stream>>>(Wv, Wvt);
  transpose_cvt_bf16<<<tg, tb, 0, stream>>>(Wo, Wot);

  dim3 gg(8, 64);  // N/128 x M/128 (M = B*S = 8192)
  const float qscale = 0.125f;  // 1/sqrt(64), folded into Q
  gemm_wmma<1, 0><<<gg, 256, 0, stream>>>((const void*)x, Wqt, bq, (void*)Qb, qscale);
  gemm_wmma<1, 0><<<gg, 256, 0, stream>>>((const void*)x, Wkt, bk, (void*)Kb, 1.0f);
  gemm_wmma<1, 1><<<gg, 256, 0, stream>>>((const void*)x, Wvt, bv, (void*)Vtb, 1.0f);

  dim3 ag(SEQ / 64, 4 * NH);
  attention_wmma<<<ag, 128, 0, stream>>>(Qb, Kb, Vtb, Ob);

  gemm_wmma<0, 2><<<gg, 256, 0, stream>>>((const void*)Ob, Wot, bo, d_out, 1.0f);
}
